// MLA_32384053411984
// MI455X (gfx1250) — matrix operand order repaired and checked by reading, not compiled, not run on hardware
//
#include <hip/hip_runtime.h>
#include <hip/hip_bf16.h>
#include <math.h>

#define B_DIM   2
#define T_DIM   2048
#define NEMBD   2048
#define LATENT  1024
#define NHEAD   16
#define HD      64
#define QKV_COLS (3 * LATENT)   // 3072

typedef __attribute__((ext_vector_type(16))) __bf16 v16bf;
typedef __attribute__((ext_vector_type(8)))  float  v8f;

// async-to-LDS builtin takes int4 pointers: addrspace(1) src, addrspace(3) dst
typedef int v4i __attribute__((vector_size(16)));
typedef __attribute__((address_space(1))) v4i* gas1_v4i;
typedef __attribute__((address_space(3))) v4i* las3_v4i;

#if defined(__HIP_DEVICE_COMPILE__) && \
    __has_builtin(__builtin_amdgcn_global_load_async_to_lds_b128) && \
    __has_builtin(__builtin_amdgcn_s_wait_asynccnt)
#define HAVE_ASYNC_LDS 1
#else
#define HAVE_ASYNC_LDS 0
#endif

static __device__ __forceinline__ v8f wmma_bf16f32(v16bf a, v16bf b, v8f c) {
  // D(f32 16x16) = A(bf16 16x32) * B(bf16 32x16) + C
  return __builtin_amdgcn_wmma_f32_16x16x32_bf16(false, a, false, b, (short)0, c,
                                                 false, false);
}

// ---------------------------------------------------------------------------
// GEMM: C[M,N](f32) = A[M,K](f32) * B[K,N](f32); bf16 WMMA, fp32 accumulate.
// Block = 256 thr (8 waves). Block tile = 256(M) x 64(N); each wave owns a
// 32x64 stripe (2 A-frags x 4 B-frags = 8 WMMA per 32-K step). The 32x64 B
// tile is staged in LDS once per block per K-step (async-to-LDS if available).
// ---------------------------------------------------------------------------
__global__ void __launch_bounds__(256) gemm_bf16_wmma(
    const float* __restrict__ A, const float* __restrict__ Bm,
    float* __restrict__ C, int M, int N, int K, int gridN) {
  __shared__ float ldsB[32 * 64];

  const int tid  = threadIdx.x;
  const int lane = tid & 31;
  const int m    = lane & 15;
  const int hi   = lane >> 4;
  const int w    = tid >> 5;
  const int bm   = blockIdx.x / gridN;
  const int bn   = blockIdx.x - bm * gridN;
  const int rb   = bm * 256 + w * 32;   // wave's 32-row stripe
  const int cn0  = bn * 64;             // block's 64-col stripe
  const int row0 = rb + m;
  const int row1 = rb + 16 + m;

  // cooperative B-tile copy: two 16-byte chunks per thread (32x64 f32 = 8KB)
  const int i0 = tid, i1 = tid + 256;
  const int r0c = i0 >> 4, c0c = (i0 & 15) << 2;
  const int r1c = i1 >> 4, c1c = (i1 & 15) << 2;

  v8f acc[2][4] = {};

  for (int k0 = 0; k0 < K; k0 += 32) {
#if HAVE_ASYNC_LDS
    __builtin_amdgcn_global_load_async_to_lds_b128(
        (gas1_v4i)(Bm + (size_t)(k0 + r0c) * N + cn0 + c0c),
        (las3_v4i)(ldsB + r0c * 64 + c0c), 0, 0);
    __builtin_amdgcn_global_load_async_to_lds_b128(
        (gas1_v4i)(Bm + (size_t)(k0 + r1c) * N + cn0 + c1c),
        (las3_v4i)(ldsB + r1c * 64 + c1c), 0, 0);
    __builtin_amdgcn_s_wait_asynccnt(0);
#else
    *(float4*)(ldsB + r0c * 64 + c0c) =
        *(const float4*)(Bm + (size_t)(k0 + r0c) * N + cn0 + c0c);
    *(float4*)(ldsB + r1c * 64 + c1c) =
        *(const float4*)(Bm + (size_t)(k0 + r1c) * N + cn0 + c1c);
#endif
    __syncthreads();

    if (k0 + 32 < K) {
      __builtin_prefetch(&A[(size_t)row0 * K + k0 + 32], 0, 0);
      __builtin_prefetch(&A[(size_t)row1 * K + k0 + 32], 0, 0);
    }

    // A fragments: lane = row, K index ka(e) = e+16*hi
    v16bf a0, a1;
#pragma unroll
    for (int e = 0; e < 16; ++e) {
      const int ka = e + 16 * hi;
      a0[e] = (__bf16)A[(size_t)row0 * K + k0 + ka];
      a1[e] = (__bf16)A[(size_t)row1 * K + k0 + ka];
    }

#pragma unroll
    for (int f = 0; f < 4; ++f) {
      v16bf b;  // B fragment from LDS: lane = column f*16+m, K = e+16*hi
#pragma unroll
      for (int e = 0; e < 16; ++e)
        b[e] = (__bf16)ldsB[(e + 16 * hi) * 64 + f * 16 + m];
      acc[0][f] = wmma_bf16f32(a0, b, acc[0][f]);
      acc[1][f] = wmma_bf16f32(a1, b, acc[1][f]);
    }
    __syncthreads();
  }

#pragma unroll
  for (int i = 0; i < 2; ++i)
#pragma unroll
    for (int f = 0; f < 4; ++f)
#pragma unroll
      for (int r = 0; r < 8; ++r)
        C[(size_t)(rb + i * 16 + r + 8 * hi) * N + cn0 + f * 16 + m] =
            acc[i][f][r];
}

// ---------------------------------------------------------------------------
// RoPE (interleaved pairs, full 64-dim head) + split qkv into head-major
// bf16 tensors q/k/v[B][H][T][64]. One thread per (b,h,t,pair).
// ---------------------------------------------------------------------------
__global__ void rope_split_kernel(const float* __restrict__ qkv,
                                  __bf16* __restrict__ qb,
                                  __bf16* __restrict__ kb,
                                  __bf16* __restrict__ vb) {
  const int i  = blockIdx.x * blockDim.x + threadIdx.x;  // ((bh)*T + t)*32 + p
  const int p  = i & 31;
  const int t  = (i >> 5) & (T_DIM - 1);
  const int bh = i >> 16;
  const int b  = bh >> 4;
  const int h  = bh & 15;

  // inv_freq = 10000^(-p/32); ang = t * inv_freq
  const float ang = (float)t * __expf(-(float)p * (9.2103403719761836f / 32.0f));
  float s, c;
  __sincosf(ang, &s, &c);

  const size_t ro   = (size_t)(b * T_DIM + t) * QKV_COLS;
  const int    colq = h * HD + 2 * p;
  const float q0 = qkv[ro + colq],              q1 = qkv[ro + colq + 1];
  const float k0 = qkv[ro + LATENT + colq],     k1 = qkv[ro + LATENT + colq + 1];
  const float v0 = qkv[ro + 2 * LATENT + colq], v1 = qkv[ro + 2 * LATENT + colq + 1];

  const size_t o = ((size_t)bh * T_DIM + t) * HD + 2 * p;
  qb[o]     = (__bf16)(q0 * c - q1 * s);
  qb[o + 1] = (__bf16)(q1 * c + q0 * s);
  kb[o]     = (__bf16)(k0 * c - k1 * s);
  kb[o + 1] = (__bf16)(k1 * c + k0 * s);
  vb[o]     = (__bf16)v0;
  vb[o + 1] = (__bf16)v1;
}

// ---------------------------------------------------------------------------
// Causal flash attention. One wave (32 threads) per (b,h, 16-query tile).
// S^T = K_tile * Q^T so queries sit in lanes (row stats need one shfl_xor 16);
// the S^T C-fragment maps 1:1 onto the P A-fragment for the PV WMMA.
// ---------------------------------------------------------------------------
__global__ void flash_attn_kernel(const __bf16* __restrict__ qb,
                                  const __bf16* __restrict__ kb,
                                  const __bf16* __restrict__ vb,
                                  float* __restrict__ y) {
  const int lane = threadIdx.x & 31;
  const int m    = lane & 15;
  const int hi   = lane >> 4;
  const int bh   = blockIdx.y;
  const int b    = bh >> 4;
  const int h    = bh & 15;
  const int q0   = blockIdx.x << 4;
  const size_t base = (size_t)bh * T_DIM * HD;

  // Q^T B-fragments (column = query = m) for dim halves [0,32) and [32,64)
  v16bf qf0, qf1;
#pragma unroll
  for (int e = 0; e < 16; ++e) {
    const size_t qi = base + (size_t)(q0 + m) * HD + e + 16 * hi;
    qf0[e] = qb[qi];
    qf1[e] = qb[qi + 32];
  }

  float m_run = -INFINITY, l_run = 0.0f;
  v8f acc0 = {}, acc1 = {}, acc2 = {}, acc3 = {};
  const float scale = 0.125f;  // 1/sqrt(64)
  const int qg = q0 + m;       // this lane's global query index

  for (int k0 = 0; k0 <= q0 + 15; k0 += 32) {
    // K-tile A-fragments (row = key): two 16-key halves x two dim chunks
    v16bf ak0d0, ak0d1, ak1d0, ak1d1;
#pragma unroll
    for (int e = 0; e < 16; ++e) {
      const int ka = e + 16 * hi;
      const size_t r0 = base + (size_t)(k0 + m) * HD;
      const size_t r1 = base + (size_t)(k0 + 16 + m) * HD;
      ak0d0[e] = kb[r0 + ka];
      ak0d1[e] = kb[r0 + 32 + ka];
      ak1d0[e] = kb[r1 + ka];
      ak1d1[e] = kb[r1 + 32 + ka];
    }
    v8f s0 = {}, s1 = {};
    s0 = wmma_bf16f32(ak0d0, qf0, s0);
    s0 = wmma_bf16f32(ak0d1, qf1, s0);
    s1 = wmma_bf16f32(ak1d0, qf0, s1);
    s1 = wmma_bf16f32(ak1d1, qf1, s1);

    // scale + causal mask; lane holds query qg, keys k0 + r + 8*hi (+16 for s1)
    float sv[16];
#pragma unroll
    for (int r = 0; r < 8; ++r) {
      const int key0 = k0 + r + 8 * hi;
      sv[r]     = (key0      <= qg) ? s0[r] * scale : -1e30f;
      sv[r + 8] = (key0 + 16 <= qg) ? s1[r] * scale : -1e30f;
    }
    float tmax = sv[0];
#pragma unroll
    for (int j = 1; j < 16; ++j) tmax = fmaxf(tmax, sv[j]);
    tmax = fmaxf(tmax, __shfl_xor(tmax, 16, 32));
    const float m_new = fmaxf(m_run, tmax);
    const float alpha = __expf(m_run - m_new);

    float pv[16];
    float lsum = 0.0f;
#pragma unroll
    for (int j = 0; j < 16; ++j) { pv[j] = __expf(sv[j] - m_new); lsum += pv[j]; }
    lsum += __shfl_xor(lsum, 16, 32);
    l_run = l_run * alpha + lsum;
    m_run = m_new;

    // rescale accumulators: row r+8*hi's alpha lives in lane (r+8*hi)
    float arow[8];
#pragma unroll
    for (int r = 0; r < 8; ++r) arow[r] = __shfl(alpha, r + 8 * hi, 32);
#pragma unroll
    for (int r = 0; r < 8; ++r) {
      acc0[r] *= arow[r]; acc1[r] *= arow[r];
      acc2[r] *= arow[r]; acc3[r] *= arow[r];
    }

    // P A-fragment: S^T C-layout element j maps exactly to A-fragment elem j
    v16bf pf;
#pragma unroll
    for (int e = 0; e < 16; ++e) pf[e] = (__bf16)pv[e];

    // V B-fragments (column = output dim chunk f*16 + m, row = key e<8 ? e+8*hi : 8+e+8*hi)
    v16bf vf0, vf1, vf2, vf3;
#pragma unroll
    for (int e = 0; e < 16; ++e) {
      const size_t vr = base + (size_t)(k0 + ((e < 8) ? (e + 8 * hi) : (8 + e + 8 * hi))) * HD;
      vf0[e] = vb[vr + m];
      vf1[e] = vb[vr + 16 + m];
      vf2[e] = vb[vr + 32 + m];
      vf3[e] = vb[vr + 48 + m];
    }
    acc0 = wmma_bf16f32(pf, vf0, acc0);
    acc1 = wmma_bf16f32(pf, vf1, acc1);
    acc2 = wmma_bf16f32(pf, vf2, acc2);
    acc3 = wmma_bf16f32(pf, vf3, acc3);
  }

  // normalize and store to y[B,T,LATENT] (fp32) for the out-projection
  float lrow[8];
#pragma unroll
  for (int r = 0; r < 8; ++r) lrow[r] = __shfl(l_run, r + 8 * hi, 32);
#pragma unroll
  for (int r = 0; r < 8; ++r) {
    const float inv = 1.0f / lrow[r];
    const size_t yrow = (size_t)(b * T_DIM + q0 + r + 8 * hi) * LATENT + h * HD;
    y[yrow + m]      = acc0[r] * inv;
    y[yrow + 16 + m] = acc1[r] * inv;
    y[yrow + 32 + m] = acc2[r] * inv;
    y[yrow + 48 + m] = acc3[r] * inv;
  }
}

// ---------------------------------------------------------------------------
extern "C" void kernel_launch(void* const* d_in, const int* in_sizes, int n_in,
                              void* d_out, int out_size, void* d_ws, size_t ws_size,
                              hipStream_t stream) {
  (void)in_sizes; (void)n_in; (void)out_size; (void)ws_size;
  const float* x      = (const float*)d_in[0];
  const float* w_attn = (const float*)d_in[1];
  const float* w_proj = (const float*)d_in[2];
  float*       out    = (float*)d_out;

  char* ws = (char*)d_ws;
  float*  qkv = (float*)ws;                                      // 4096*3072 f32
  float*  y   = (float*)(ws + (size_t)4096 * 3072 * 4);          // 4096*1024 f32
  __bf16* qb  = (__bf16*)(ws + (size_t)4096 * 3072 * 4 + (size_t)4096 * 1024 * 4);
  __bf16* kb  = qb + (size_t)B_DIM * NHEAD * T_DIM * HD;
  __bf16* vb  = kb + (size_t)B_DIM * NHEAD * T_DIM * HD;

  {  // 1) qkv = x @ w_attn : (4096x2048)@(2048x3072)
    const int M = B_DIM * T_DIM, N = QKV_COLS, K = NEMBD;
    const int gridN = N / 64;                       // 48
    gemm_bf16_wmma<<<(M / 256) * gridN, 256, 0, stream>>>(x, w_attn, qkv, M, N, K,
                                                          gridN);
  }
  {  // 2) RoPE + head split to bf16
    const int total = B_DIM * NHEAD * T_DIM * 32;   // 2,097,152
    rope_split_kernel<<<total / 256, 256, 0, stream>>>(qkv, qb, kb, vb);
  }
  // 3) causal flash attention (one wave per 16-query tile per head)
  flash_attn_kernel<<<dim3(T_DIM / 16, B_DIM * NHEAD), 32, 0, stream>>>(qb, kb, vb, y);
  {  // 4) out = y @ w_proj : (4096x1024)@(1024x2048)
    const int M = B_DIM * T_DIM, N = NEMBD, K = LATENT;
    const int gridN = N / 64;                       // 32
    gemm_bf16_wmma<<<(M / 256) * gridN, 256, 0, stream>>>(y, w_proj, out, M, N, K,
                                                          gridN);
  }
}
